// MultiHeadSelfAttention2d_11931419148694
// MI455X (gfx1250) — matrix operand order repaired and checked by reading, not compiled, not run on hardware
//
#include <hip/hip_runtime.h>
#include <hip/hip_bf16.h>

// MI455X (gfx1250) fused multi-head self-attention.
// attn output (805 MB) dominates HBM -> fuse scores/softmax/attn-write/P@V so
// attn is written once. Probability panel kept in f16 in LDS (halves LDS
// traffic, contiguous ds_load_b128 A-tiles). V tiles staged into LDS by the
// Tensor Data Mover (double buffered, s_wait_tensorcnt), overlapped with the
// HBM-bound attn write. All matmuls: v_wmma_f32_16x16x32_f16.

typedef __attribute__((ext_vector_type(16))) _Float16 v16h;
typedef __attribute__((ext_vector_type(8)))  _Float16 v8h;
typedef __attribute__((ext_vector_type(8)))  float    v8f;
typedef unsigned int u32x4 __attribute__((ext_vector_type(4)));
typedef int          i32x4 __attribute__((ext_vector_type(4)));
typedef int          i32x8 __attribute__((ext_vector_type(8)));

#define NQ 4096        // S*L tokens
#define DM 768         // model dim
#define EE 2304        // 3*DM
#define NH 12          // heads
#define DH 64          // head dim

#define PH_STRIDE 4104                       // f16 panel row stride (halves)
#define PH_BYTES  (16 * PH_STRIDE * 2)       // 131328
#define VS_ROWSTR 48                         // staged V row stride (halves) = 96B
#define VS_BUF_BYTES (64 * VS_ROWSTR * 2)    // 6144 per buffer
#define VS_BYTES  (8 * 2 * VS_BUF_BYTES)     // 98304
#define OR_BYTES  (8 * 16 * 64 * 4)          // 32768
#define RR_BYTES  (16 * 16 * 4)              // 1024
#define SMEM_BYTES (PH_BYTES + VS_BYTES + OR_BYTES + RR_BYTES) // 263424 < 320KB

__device__ __forceinline__ v8f wmma_f16(v16h a, v16h b, v8f c) {
  return __builtin_amdgcn_wmma_f32_16x16x32_f16(
      /*neg_a=*/false, a, /*neg_b=*/false, b,
      /*c_mod=*/(short)0, c, /*reuse_a=*/false, /*reuse_b=*/false);
}

// ---------------------------------------------------------------- convert
__global__ void f32_to_f16_kernel(const float* __restrict__ src,
                                  _Float16* __restrict__ dst, int n) {
  int i = blockIdx.x * blockDim.x + threadIdx.x;
  if (i < n) dst[i] = (_Float16)src[i];
}

// ---------------------------------------------------------------- QKV GEMM
// One wave = one 16(n) x 16(e) tile of qkv = x @ W^T + b, K=768.
// Writes q (scaled), k as f16 [h][n][64]; v transposed f16 [h][64][n].
__global__ __launch_bounds__(256)
void qkv_proj_kernel(const _Float16* __restrict__ xh,
                     const _Float16* __restrict__ wh,
                     const float* __restrict__ bias,
                     _Float16* __restrict__ qh,
                     _Float16* __restrict__ kh,
                     _Float16* __restrict__ vt) {
  const int wave = threadIdx.x >> 5;
  const int lane = threadIdx.x & 31;
  const int tile = blockIdx.x * 8 + wave;
  const int n0 = (tile / (EE / 16)) * 16;
  const int e0 = (tile % (EE / 16)) * 16;
  const int row = lane & 15;
  const int g   = lane >> 4;
  const int col = lane & 15;

  v8f c;
  {
    float b = bias[e0 + col];
#pragma unroll
    for (int r = 0; r < 8; ++r) c[r] = b;
  }

  const _Float16* xrow = xh + (size_t)(n0 + row) * DM;
  const _Float16* wrow = wh + (size_t)(e0 + col) * DM;
  for (int kc = 0; kc < DM; kc += 32) {
    v8h alo = *(const v8h*)(xrow + kc + g * 16);
    v8h ahi = *(const v8h*)(xrow + kc + g * 16 + 8);
    v16h a;
#pragma unroll
    for (int i = 0; i < 8; ++i) { a[i] = alo[i]; a[i + 8] = ahi[i]; }
    v16h bt = *(const v16h*)(wrow + kc + g * 16);
    c = wmma_f16(a, bt, c);
  }

  const int seg  = e0 / DM;
  const int eo   = e0 % DM;
  const int head = eo / DH;
  const int dim0 = eo % DH;
#pragma unroll
  for (int r = 0; r < 8; ++r) {
    int m = n0 + r + g * 8;
    float val = c[r];
    if (seg == 0)      qh[(size_t)head * NQ * DH + (size_t)m * DH + dim0 + col] =
                           (_Float16)(val * 0.125f);
    else if (seg == 1) kh[(size_t)head * NQ * DH + (size_t)m * DH + dim0 + col] =
                           (_Float16)val;
    else               vt[(size_t)head * DH * NQ + (size_t)(dim0 + col) * NQ + m] =
                           (_Float16)val;
  }
}

// ---------------------------------------------------------------- attention
__global__ __launch_bounds__(256)
void attn_kernel(const _Float16* __restrict__ qh,
                 const _Float16* __restrict__ kh,
                 const _Float16* __restrict__ vt,
                 const unsigned char* __restrict__ pmask,
                 float* __restrict__ out,
                 float* __restrict__ attn) {
  extern __shared__ char smem[];
  _Float16* ph     = (_Float16*)smem;                              // [16][PH_STRIDE] f16
  _Float16* vstage = (_Float16*)(smem + PH_BYTES);                 // [8][2][64][48]
  float*    outred = (float*)(smem + PH_BYTES + VS_BYTES);         // [8][16][64]
  float*    rowred = (float*)(smem + PH_BYTES + VS_BYTES + OR_BYTES); // [16][16]

  const int h   = blockIdx.x / (NQ / 16);
  const int qb  = blockIdx.x % (NQ / 16);
  const int n0  = qb * 16;
  const int tid = threadIdx.x;
  const int wave = tid >> 5;
  const int lane = tid & 31;
  const int row = lane & 15, g = lane >> 4, col = lane & 15;

  // Q A-tiles, persistent in VGPRs (dh=64 -> two 16x32 tiles)
  v16h a0, a1;
  {
    const _Float16* qrow = qh + (size_t)h * NQ * DH + (size_t)(n0 + row) * DH;
    v8h l0 = *(const v8h*)(qrow + g * 16);
    v8h h0 = *(const v8h*)(qrow + g * 16 + 8);
    v8h l1 = *(const v8h*)(qrow + 32 + g * 16);
    v8h h1 = *(const v8h*)(qrow + 40 + g * 16);
#pragma unroll
    for (int i = 0; i < 8; ++i) {
      a0[i] = l0[i]; a0[i + 8] = h0[i];
      a1[i] = l1[i]; a1[i + 8] = h1[i];
    }
  }

  // ---- TDM issue helper: stage a 32-key x 64-dim tile of V^T into LDS
  const int kstart = wave * (NQ / 8);
  const unsigned lds_base = (unsigned)PH_BYTES + (unsigned)(wave * 2) * VS_BUF_BYTES;
  auto issue_tdm = [&](int chunk, int buf) {
    unsigned long long ga = (unsigned long long)(uintptr_t)
        (vt + (size_t)h * DH * NQ + (size_t)(kstart + chunk * 32));
    // D# group0: count=1 | lds_addr | global_addr | type=2
    u32x4 g0 = { 1u,
                 lds_base + (unsigned)buf * VS_BUF_BYTES,
                 (unsigned)(ga & 0xffffffffu),
                 (unsigned)((ga >> 32) & 0x01ffffffu) | 0x80000000u };
    // D# group1: data_size=2B, pad_enable, pad_interval=16dw, pad_amount=8dw,
    // tensor_dim0=4096, tensor_dim1=64, tile_dim0=32, tile_dim1=64,
    // tensor_dim0_stride=4096
    i32x8 g1 = { 0x0ED10000, 0x10000000, 0x00400000, 0x00200000,
                 64, 4096, 0, 0 };
    i32x4 gz = { 0, 0, 0, 0 };
#if __clang_major__ >= 23
    i32x8 gz8 = { 0, 0, 0, 0, 0, 0, 0, 0 };
    __builtin_amdgcn_tensor_load_to_lds(g0, g1, gz, gz, gz8, 0);
#else
    __builtin_amdgcn_tensor_load_to_lds(g0, g1, gz, gz, 0);
#endif
  };

  // ---- scores: 8 waves tile 256 key-chunks of 16; panel stored as f16
  for (int kc = wave; kc < NQ / 16; kc += 8) {
    const int key0 = kc * 16;
    const _Float16* krow = kh + (size_t)h * NQ * DH + (size_t)(key0 + col) * DH;
    __builtin_prefetch((const void*)(krow + 8 * 16 * DH), 0, 0);
    v16h b0 = *(const v16h*)(krow + g * 16);
    v16h b1 = *(const v16h*)(krow + 32 + g * 16);
    v8f s = {};
    s = wmma_f16(a0, b0, s);
    s = wmma_f16(a1, b1, s);
#pragma unroll
    for (int r = 0; r < 8; ++r)
      ph[(r + g * 8) * PH_STRIDE + key0 + col] = (_Float16)s[r];
  }
  __syncthreads();

  // ---- additive key-padding mask
  for (int c2 = tid; c2 < NQ; c2 += 256) {
    if (pmask[c2]) {
#pragma unroll
      for (int r = 0; r < 16; ++r)
        ph[r * PH_STRIDE + c2] = (_Float16)(-__builtin_inff());
    }
  }
  __syncthreads();

  // ---- softmax over each of the 16 rows (16 threads per row)
  {
    const int r = tid >> 4, sub = tid & 15;
    float m = -__builtin_inff();
    for (int c2 = sub; c2 < NQ; c2 += 16)
      m = fmaxf(m, (float)ph[r * PH_STRIDE + c2]);
    rowred[r * 16 + sub] = m;
    __syncthreads();
    float mm = -__builtin_inff();
#pragma unroll
    for (int i = 0; i < 16; ++i) mm = fmaxf(mm, rowred[r * 16 + i]);
    __syncthreads();
    float sum = 0.f;
    for (int c2 = sub; c2 < NQ; c2 += 16) {
      float e = __expf((float)ph[r * PH_STRIDE + c2] - mm);
      ph[r * PH_STRIDE + c2] = (_Float16)e;
      sum += e;
    }
    rowred[r * 16 + sub] = sum;
    __syncthreads();
    float tot = 0.f;
#pragma unroll
    for (int i = 0; i < 16; ++i) tot += rowred[r * 16 + i];
    float inv = 1.f / tot;
    for (int c2 = sub; c2 < NQ; c2 += 16)
      ph[r * PH_STRIDE + c2] = (_Float16)((float)ph[r * PH_STRIDE + c2] * inv);
  }
  __syncthreads();

  // ---- kick off first V tile DMA, then overlap with the HBM-bound attn write
  issue_tdm(0, 0);
  {
    float* ap = attn + (size_t)h * NQ * NQ + (size_t)n0 * NQ;
    for (int r = 0; r < 16; ++r)
      for (int c2 = tid; c2 < NQ; c2 += 256)
        ap[(size_t)r * NQ + c2] = (float)ph[r * PH_STRIDE + c2];
  }

  // ---- out(16x64) = P(16x4096) @ V(4096x64); wave owns full dh over K/8
  {
    v8f acc[4] = { {}, {}, {}, {} };
    for (int i = 0; i < 16; ++i) {
      if (i < 15) {
        asm volatile("s_wait_dscnt 0x0" ::: "memory"); // prior reads of target buf done
        issue_tdm(i + 1, (i + 1) & 1);
        __builtin_amdgcn_s_wait_tensorcnt(1);
      } else {
        __builtin_amdgcn_s_wait_tensorcnt(0);
      }
      const int kk = kstart + i * 32;
      // A tile: contiguous f16 LDS loads (two 16B ds_load_b128 per lane)
      const _Float16* prow = ph + row * PH_STRIDE + kk;
      v8h plo = *(const v8h*)(prow + g * 16);
      v8h phi = *(const v8h*)(prow + g * 16 + 8);
      v16h pa;
#pragma unroll
      for (int t = 0; t < 8; ++t) { pa[t] = plo[t]; pa[t + 8] = phi[t]; }
      // B tiles from the TDM-staged buffer [dim][48]
      const _Float16* vb = vstage + (size_t)(wave * 2 + (i & 1)) * (64 * VS_ROWSTR);
#pragma unroll
      for (int dt = 0; dt < 4; ++dt) {
        const _Float16* vr = vb + (dt * 16 + col) * VS_ROWSTR;
        v8h bl = *(const v8h*)(vr + g * 16);
        v8h bh = *(const v8h*)(vr + g * 16 + 8);
        v16h b;
#pragma unroll
        for (int t = 0; t < 8; ++t) { b[t] = bl[t]; b[t + 8] = bh[t]; }
        acc[dt] = wmma_f16(pa, b, acc[dt]);
      }
    }
#pragma unroll
    for (int dt = 0; dt < 4; ++dt)
#pragma unroll
      for (int r = 0; r < 8; ++r)
        outred[wave * 1024 + (r + g * 8) * 64 + dt * 16 + col] = acc[dt][r];
  }
  __syncthreads();

  // ---- 8-way reduce over K slices, write out [n][h*64+d]
  for (int idx = tid; idx < 16 * DH; idx += 256) {
    int r = idx >> 6, d = idx & 63;
    float s = 0.f;
#pragma unroll
    for (int w = 0; w < 8; ++w) s += outred[w * 1024 + r * 64 + d];
    out[(size_t)(n0 + r) * DM + h * DH + d] = s;
  }
}

// ---------------------------------------------------------------- launch
extern "C" void kernel_launch(void* const* d_in, const int* in_sizes, int n_in,
                              void* d_out, int out_size, void* d_ws, size_t ws_size,
                              hipStream_t stream) {
  const float*         x     = (const float*)d_in[0];
  const unsigned char* pmask = (const unsigned char*)d_in[1];
  const float*         W     = (const float*)d_in[2];
  const float*         b     = (const float*)d_in[3];

  float* out  = (float*)d_out;
  float* attn = out + (size_t)NQ * DM;   // tuple: (out, attn) flat

  _Float16* xh = (_Float16*)d_ws;
  _Float16* wh = xh + (size_t)NQ * DM;
  _Float16* qh = wh + (size_t)EE * DM;
  _Float16* kh = qh + (size_t)NH * NQ * DH;
  _Float16* vt = kh + (size_t)NH * NQ * DH;

  f32_to_f16_kernel<<<(NQ * DM + 255) / 256, 256, 0, stream>>>(x, xh, NQ * DM);
  f32_to_f16_kernel<<<(EE * DM + 255) / 256, 256, 0, stream>>>(W, wh, EE * DM);

  qkv_proj_kernel<<<(NQ / 16) * (EE / 16) / 8, 256, 0, stream>>>(
      xh, wh, b, qh, kh, vt);

  attn_kernel<<<NH * (NQ / 16), 256, SMEM_BYTES, stream>>>(
      qh, kh, vt, pmask, out, attn);
}
